// MultiHeadAttention_31679678775974
// MI455X (gfx1250) — matrix operand order repaired and checked by reading, not compiled, not run on hardware
//
#include <hip/hip_runtime.h>
#include <hip/hip_bf16.h>

typedef __attribute__((ext_vector_type(16))) _Float16 v16h;
typedef __attribute__((ext_vector_type(8)))  _Float16 v8h;
typedef __attribute__((ext_vector_type(8)))  float    v8f;
typedef __attribute__((ext_vector_type(4)))  unsigned int u32x4;
typedef __attribute__((ext_vector_type(8)))  int          i32x8;
typedef __attribute__((ext_vector_type(4)))  int          i32x4;

// address-space qualified vector types for the async-to-LDS builtin
typedef __attribute__((address_space(1))) i32x4 as1_i32x4;
typedef __attribute__((address_space(3))) i32x4 as3_i32x4;

union Frag16 { v16h v; v8h h[2]; };

__device__ __forceinline__ v8f wmma16(const Frag16& a, const Frag16& b, v8f c) {
    return __builtin_amdgcn_wmma_f32_16x16x32_f16(
        /*neg_a=*/false, a.v, /*neg_b=*/false, b.v,
        /*c_mod=*/(short)0, c, /*reuse_a=*/false, /*reuse_b=*/false);
}

// ----------------------------------------------------- async / TDM support ---
#if __has_builtin(__builtin_amdgcn_global_load_async_to_lds_b128)
#define HAVE_ASYNC 1
#else
#define HAVE_ASYNC 0
#endif

#if __has_builtin(__builtin_amdgcn_tensor_load_to_lds) && __has_builtin(__builtin_amdgcn_s_wait_tensorcnt)
#define HAVE_TDM 1
#else
#define HAVE_TDM 0
#endif

#if HAVE_ASYNC
__device__ __forceinline__ void async_copy_b128(const _Float16* g, _Float16* l) {
    __builtin_amdgcn_global_load_async_to_lds_b128(
        (as1_i32x4*)(uintptr_t)g,
        (as3_i32x4*)(unsigned)(uintptr_t)l,
        /*offset=*/0, /*cpol=*/0);
}
__device__ __forceinline__ void wait_async_all() {
#if __has_builtin(__builtin_amdgcn_s_wait_asynccnt)
    __builtin_amdgcn_s_wait_asynccnt(0);
#else
    asm volatile("s_wait_asynccnt 0x0" ::: "memory");
#endif
}
#endif

#if HAVE_TDM
// 1-D TDM load of 2048 f16 elements (32 rows x 64 halfs) into LDS, with the
// pad feature inserting 16B after every 128B row -> lands as lK[32][72].
// D# bitfields per CDNA5 ISA sect. 8.3/8.4.
__device__ __forceinline__ void tdm_load_ktile(const _Float16* gsrc,
                                               unsigned lds_byte_addr) {
    unsigned long long ga = (unsigned long long)(uintptr_t)gsrc;
    u32x4 g0;
    g0[0] = 1u;                                         // count=1, user mode
    g0[1] = lds_byte_addr;                              // lds_addr
    g0[2] = (unsigned)(ga & 0xFFFFFFFFu);               // global_addr[31:0]
    g0[3] = (unsigned)((ga >> 32) & 0x1FFFFFFu)         // global_addr[56:32]
          | (2u << 30);                                 // type=2 ("image")
    const unsigned DIM0 = 2048u;                        // elements (2B each)
    i32x8 g1;
    g1[0] = (int)((1u << 16)        // data_size=1 -> 2 bytes
                | (1u << 20)        // pad_enable
                | (4u << 22)        // pad_interval: code 4 = 32 DWORDs (128B)
                | (3u << 25));      // pad_amount:   code 3 = 4 DWORDs (16B)
    g1[1] = (int)((DIM0 & 0xFFFFu) << 16);              // tensor_dim0 lo16
    g1[2] = (int)((DIM0 >> 16) | (1u << 16));           // dim0 hi16 | tensor_dim1=1
    g1[3] = (int)(DIM0 << 16);                          // dim1 hi16=0 | tile_dim0
    g1[4] = 0;                                          // tile_dim1=0, tile_dim2=0
    g1[5] = (int)DIM0;                                  // tensor_dim0_stride lo32
    g1[6] = (int)((DIM0 & 0xFFFFu) << 16);              // stride0 hi16=0 | stride1 lo16
    g1[7] = (int)(DIM0 >> 16);                          // stride1 hi
    i32x4 gz4 = {0, 0, 0, 0};                           // <=2D: groups 2/3 unused
    i32x8 gz8 = {0, 0, 0, 0, 0, 0, 0, 0};               // 5th operand (clang-23 form)
    __builtin_amdgcn_tensor_load_to_lds(g0, g1, gz4, gz4, gz8, /*cpol=*/0);
}
#endif

// ---------------------------------------------------------------- convert ---
__global__ void mha_conv_f32_f16(const float* __restrict__ src,
                                 _Float16* __restrict__ dst, int n) {
    int i = blockIdx.x * blockDim.x + threadIdx.x;
    if (i < n) dst[i] = (_Float16)src[i];
}

// ------------------------------------------------------------------- GEMM ---
// Double-buffered: stage tile k+1 (async DMA) while WMMAs run on tile k.
#define BM 128
#define BN 128
#define BK 32

__global__ __launch_bounds__(256) void mha_gemm_f16_wmma(
    const _Float16* __restrict__ A, const _Float16* __restrict__ W,
    const float* __restrict__ bias, float* __restrict__ C,
    int Md, int Nd, int Kd) {
    __shared__ alignas(16) _Float16 lA[2][BM][48];   // [m][k], 96B stride
    __shared__ alignas(16) _Float16 lB[2][BN][48];   // [n][k], W transposed

    const int tid  = threadIdx.x;
    const int wv   = tid >> 5;
    const int lane = tid & 31;
    const int hs   = lane >> 4;
    const int ln   = lane & 15;
    const int wm   = wv >> 2;
    const int wn   = wv & 3;
    const int m0   = blockIdx.y * BM;
    const int n0   = blockIdx.x * BN;

    const int arow = tid >> 1, aseg = (tid & 1) * 16;   // A staging slot
    const int wk   = tid >> 3, nseg = (tid & 7) * 16;   // W staging slot

    auto stage = [&](_Float16 (*dA)[48], _Float16 (*dB)[48], int k0) {
        const _Float16* ga = A + (size_t)(m0 + arow) * Kd + k0 + aseg;
#if HAVE_ASYNC
        async_copy_b128(ga,     &dA[arow][aseg]);
        async_copy_b128(ga + 8, &dA[arow][aseg + 8]);
#else
        v8h x0 = *(const v8h*)ga;
        v8h x1 = *(const v8h*)(ga + 8);
        *(v8h*)&dA[arow][aseg]     = x0;
        *(v8h*)&dA[arow][aseg + 8] = x1;
#endif
        const _Float16* gw = W + (size_t)(k0 + wk) * Nd + n0 + nseg;
        v8h w0 = *(const v8h*)gw;
        v8h w1 = *(const v8h*)(gw + 8);
#pragma unroll
        for (int j = 0; j < 8; j++) dB[nseg + j][wk]     = w0[j];
#pragma unroll
        for (int j = 0; j < 8; j++) dB[nseg + 8 + j][wk] = w1[j];
    };

    v8f acc[4][2];
    v8f vzero = {0.f,0.f,0.f,0.f,0.f,0.f,0.f,0.f};
#pragma unroll
    for (int i = 0; i < 4; i++)
#pragma unroll
        for (int j = 0; j < 2; j++) acc[i][j] = vzero;

    _Float16 (*aCur)[48] = lA[0], (*aNxt)[48] = lA[1];
    _Float16 (*bCur)[48] = lB[0], (*bNxt)[48] = lB[1];

    stage(aCur, bCur, 0);
#if HAVE_ASYNC
    wait_async_all();
#endif
    __syncthreads();

    for (int k0 = 0; k0 < Kd; k0 += BK) {
        if (k0 + BK < Kd) stage(aNxt, bNxt, k0 + BK);       // overlap with WMMA
        if (k0 + 2 * BK < Kd) {
            __builtin_prefetch(A + (size_t)(m0 + arow) * Kd + k0 + 2 * BK, 0, 3);
            __builtin_prefetch(W + (size_t)(k0 + 2 * BK + wk) * Nd + n0 + nseg, 0, 3);
        }

        Frag16 af[4], bf[2];
#pragma unroll
        for (int tm = 0; tm < 4; tm++) {
            int r = wm * 64 + tm * 16 + ln;
            af[tm].h[0] = *(const v8h*)&aCur[r][hs * 8];
            af[tm].h[1] = *(const v8h*)&aCur[r][16 + hs * 8];
        }
#pragma unroll
        for (int tn = 0; tn < 2; tn++) {
            int n = wn * 32 + tn * 16 + ln;
            bf[tn].h[0] = *(const v8h*)&bCur[n][hs * 8];
            bf[tn].h[1] = *(const v8h*)&bCur[n][16 + hs * 8];
        }
#pragma unroll
        for (int tm = 0; tm < 4; tm++)
#pragma unroll
            for (int tn = 0; tn < 2; tn++)
                acc[tm][tn] = wmma16(af[tm], bf[tn], acc[tm][tn]);

#if HAVE_ASYNC
        wait_async_all();        // drain next-tile DMA issued this iteration
#endif
        __syncthreads();         // single barrier per iteration
        _Float16 (*tA)[48] = aCur; aCur = aNxt; aNxt = tA;
        _Float16 (*tB)[48] = bCur; bCur = bNxt; bNxt = tB;
    }

#pragma unroll
    for (int tm = 0; tm < 4; tm++)
#pragma unroll
        for (int tn = 0; tn < 2; tn++) {
            int gn   = n0 + wn * 32 + tn * 16 + ln;
            float bv = bias ? bias[gn] : 0.0f;
#pragma unroll
            for (int r = 0; r < 8; r++) {
                int gm = m0 + wm * 64 + tm * 16 + r + 8 * hs;
                C[(size_t)gm * Nd + gn] = acc[tm][tn][r] + bv;
            }
        }
}

// ------------------------------------------------------- RoPE + head split ---
__global__ void mha_rope_split(const float* __restrict__ qkv,
                               _Float16* __restrict__ qf,
                               _Float16* __restrict__ kf,
                               _Float16* __restrict__ vf) {
    int idx = blockIdx.x * blockDim.x + threadIdx.x;   // B*S*H*D = 4M
    int d = idx & 63;
    int h = (idx >> 6) & 15;
    int s = (idx >> 10) & 2047;
    int b = idx >> 21;
    size_t base = (size_t)(b * 2048 + s) * 3072;
    int e = h * 64 + d;
    float q = qkv[base + e];
    float k = qkv[base + 1024 + e];
    float v = qkv[base + 2048 + e];
    int   i   = d & 31;
    float ang = (float)s * __expf(-0.2878231691f * (float)i);  // 10000^(-i/32)
    float sn, cs;
    __sincosf(ang, &sn, &cs);
    int   partner = (d < 32) ? d + 32 : d - 32;
    float sgn     = (d < 32) ? -1.0f : 1.0f;
    float q2 = qkv[base + h * 64 + partner];
    float k2 = qkv[base + 1024 + h * 64 + partner];
    float qr = q * cs + sgn * q2 * sn;
    float kr = k * cs + sgn * k2 * sn;
    size_t o = ((size_t)(b * 16 + h) * 2048 + s) * 64 + d;
    qf[o] = (_Float16)(qr * 0.125f);   // 1/sqrt(64)
    kf[o] = (_Float16)kr;
    vf[o] = (_Float16)v;
}

// --------------------------------------------------------- flash attention ---
// Double-buffered K/V: TDM for kv-block i+1 overlaps QK^T + softmax + PV of i.
__global__ __launch_bounds__(256) void mha_flash_attn(
    const _Float16* __restrict__ qf, const _Float16* __restrict__ kf,
    const _Float16* __restrict__ vf, _Float16* __restrict__ of) {
    __shared__ alignas(16) _Float16 lK[2][32][72];    // [kv][d]
    __shared__ alignas(16) _Float16 lV[2][64][48];    // [d][kv]
    __shared__ alignas(16) _Float16 lP[8][16][48];    // per-wave P C->A relayout

    const int tid  = threadIdx.x;
    const int wv   = tid >> 5;
    const int lane = tid & 31;
    const int hs   = lane >> 4;
    const int ln   = lane & 15;
    const int bh   = blockIdx.y;            // b*16+h
    const int q0   = blockIdx.x * 128;
    const size_t headBase = (size_t)bh * 2048 * 64;

    const int skey = tid >> 3, sd8 = (tid & 7) * 8;   // K/V staging slot

    auto stageKV = [&](_Float16 (*dK)[72], _Float16 (*dV)[48], int kv0) {
#if HAVE_TDM
        if (wv == 0)
            tdm_load_ktile(kf + headBase + (size_t)kv0 * 64,
                           (unsigned)(uintptr_t)&dK[0][0]);
#else
        const _Float16* gk = kf + headBase + (size_t)(kv0 + skey) * 64 + sd8;
#if HAVE_ASYNC
        async_copy_b128(gk, &dK[skey][sd8]);
#else
        *(v8h*)&dK[skey][sd8] = *(const v8h*)gk;
#endif
#endif
        v8h vvv = *(const v8h*)(vf + headBase + (size_t)(kv0 + skey) * 64 + sd8);
#pragma unroll
        for (int j = 0; j < 8; j++) dV[sd8 + j][skey] = vvv[j];
    };

    auto waitStage = [&]() {
#if HAVE_TDM
        if (wv == 0) __builtin_amdgcn_s_wait_tensorcnt(0);
#elif HAVE_ASYNC
        wait_async_all();
#endif
    };

    Frag16 qfr[2];
    {
        const _Float16* qp = qf + headBase + (size_t)(q0 + wv * 16 + ln) * 64;
#pragma unroll
        for (int c = 0; c < 2; c++) {
            qfr[c].h[0] = *(const v8h*)(qp + c * 32 + hs * 16);
            qfr[c].h[1] = *(const v8h*)(qp + c * 32 + hs * 16 + 8);
        }
    }

    float mcur[8], lcur[8];
    v8f   o[4];
    v8f   vzero = {0.f,0.f,0.f,0.f,0.f,0.f,0.f,0.f};
#pragma unroll
    for (int r = 0; r < 8; r++) { mcur[r] = -1e30f; lcur[r] = 0.0f; }
#pragma unroll
    for (int dt = 0; dt < 4; dt++) o[dt] = vzero;

    _Float16 (*kCur)[72] = lK[0], (*kNxt)[72] = lK[1];
    _Float16 (*vCur)[48] = lV[0], (*vNxt)[48] = lV[1];

    const int kv_end = (2048 < q0 + 128) ? 2048 : (q0 + 128);

    stageKV(kCur, vCur, 0);
    waitStage();
    __syncthreads();

    for (int kv0 = 0; kv0 < kv_end; kv0 += 32) {
        if (kv0 + 32 < kv_end) stageKV(kNxt, vNxt, kv0 + 32);  // overlap DMA

        // S = Q @ K^T
        v8f s[2];
#pragma unroll
        for (int tn = 0; tn < 2; tn++) {
            int key = tn * 16 + ln;
            Frag16 b0, b1;
            b0.h[0] = *(const v8h*)&kCur[key][hs * 16];
            b0.h[1] = *(const v8h*)&kCur[key][hs * 16 + 8];
            b1.h[0] = *(const v8h*)&kCur[key][32 + hs * 16];
            b1.h[1] = *(const v8h*)&kCur[key][32 + hs * 16 + 8];
            v8f t = vzero;
            t = wmma16(qfr[0], b0, t);
            t = wmma16(qfr[1], b1, t);
            s[tn] = t;
        }

        // online softmax
#pragma unroll
        for (int r = 0; r < 8; r++) {
            int   qi  = q0 + wv * 16 + r + 8 * hs;
            int   kn0 = kv0 + ln;
            int   kn1 = kv0 + 16 + ln;
            float s0  = (kn0 <= qi) ? s[0][r] : -1e30f;
            float s1  = (kn1 <= qi) ? s[1][r] : -1e30f;
            float mx  = fmaxf(s0, s1);
#pragma unroll
            for (int off = 1; off < 16; off <<= 1)
                mx = fmaxf(mx, __shfl_xor(mx, off, 32));
            float mnew  = fmaxf(mcur[r], mx);
            float alpha = __expf(mcur[r] - mnew);
            float p0 = (kn0 <= qi) ? __expf(s0 - mnew) : 0.0f;
            float p1 = (kn1 <= qi) ? __expf(s1 - mnew) : 0.0f;
            float rs = p0 + p1;
#pragma unroll
            for (int off = 1; off < 16; off <<= 1)
                rs += __shfl_xor(rs, off, 32);
            lcur[r] = lcur[r] * alpha + rs;
            mcur[r] = mnew;
#pragma unroll
            for (int dt = 0; dt < 4; dt++) o[dt][r] *= alpha;
            lP[wv][r + 8 * hs][ln]      = (_Float16)p0;
            lP[wv][r + 8 * hs][16 + ln] = (_Float16)p1;
        }

        // O += P @ V
        Frag16 pf;
        pf.h[0] = *(const v8h*)&lP[wv][ln][hs * 16];
        pf.h[1] = *(const v8h*)&lP[wv][ln][hs * 16 + 8];
#pragma unroll
        for (int dt = 0; dt < 4; dt++) {
            int d = dt * 16 + ln;
            Frag16 bf;
            bf.h[0] = *(const v8h*)&vCur[d][hs * 16];
            bf.h[1] = *(const v8h*)&vCur[d][hs * 16 + 8];
            o[dt] = wmma16(pf, bf, o[dt]);
        }

        waitStage();             // drain next-tile TDM/async DMA
        __syncthreads();         // single barrier per iteration
        _Float16 (*tK)[72] = kCur; kCur = kNxt; kNxt = tK;
        _Float16 (*tV)[48] = vCur; vCur = vNxt; vNxt = tV;
    }

    // epilogue: O/l -> attn_out f16 [B,S,H,D]
    int b = bh >> 4, h = bh & 15;
#pragma unroll
    for (int dt = 0; dt < 4; dt++) {
        int d = dt * 16 + ln;
#pragma unroll
        for (int r = 0; r < 8; r++) {
            int   qi  = q0 + wv * 16 + r + 8 * hs;
            float val = o[dt][r] / lcur[r];
            of[((size_t)(b * 2048 + qi) * 16 + h) * 64 + d] = (_Float16)val;
        }
    }
}

// ------------------------------------------------------------------ launch ---
extern "C" void kernel_launch(void* const* d_in, const int* in_sizes, int n_in,
                              void* d_out, int out_size, void* d_ws, size_t ws_size,
                              hipStream_t stream) {
    const float* x     = (const float*)d_in[0];   // [2,2048,1024]
    const float* w_qkv = (const float*)d_in[1];   // [1024,3072]
    const float* b_qkv = (const float*)d_in[2];   // [3072]
    const float* w_o   = (const float*)d_in[3];   // [1024,1024]
    const float* b_o   = (const float*)d_in[4];   // [1024]
    float* out = (float*)d_out;                   // [2,2048,1024] f32

    char* ws = (char*)d_ws;
    _Float16* x_h  = (_Float16*)(ws);                          //  8 MB
    _Float16* wq_h = (_Float16*)(ws + ((size_t)8  << 20));     //  6 MB
    _Float16* wo_h = (_Float16*)(ws + ((size_t)14 << 20));     //  2 MB
    float*    qkv  = (float*)   (ws + ((size_t)16 << 20));     // 48 MB
    _Float16* q_h  = (_Float16*)(ws + ((size_t)64 << 20));     //  8 MB
    _Float16* k_h  = (_Float16*)(ws + ((size_t)72 << 20));     //  8 MB
    _Float16* v_h  = (_Float16*)(ws + ((size_t)80 << 20));     //  8 MB
    _Float16* at_h = (_Float16*)(ws + ((size_t)88 << 20));     //  8 MB (ends 96 MB)

    const int NX = 2 * 2048 * 1024;      // 4,194,304
    const int NW = 1024 * 3072;          // 3,145,728
    const int NO = 1024 * 1024;          // 1,048,576

    mha_conv_f32_f16<<<NX / 256, 256, 0, stream>>>(x, x_h, NX);
    mha_conv_f32_f16<<<NW / 256, 256, 0, stream>>>(w_qkv, wq_h, NW);
    mha_conv_f32_f16<<<NO / 256, 256, 0, stream>>>(w_o, wo_h, NO);

    mha_gemm_f16_wmma<<<dim3(3072 / BN, 4096 / BM), 256, 0, stream>>>(
        x_h, wq_h, b_qkv, qkv, 4096, 3072, 1024);

    mha_rope_split<<<NX / 256, 256, 0, stream>>>(qkv, q_h, k_h, v_h);

    mha_flash_attn<<<dim3(16, 32), 256, 0, stream>>>(q_h, k_h, v_h, at_h);

    mha_gemm_f16_wmma<<<dim3(1024 / BN, 4096 / BM), 256, 0, stream>>>(
        at_h, wo_h, b_o, out, 4096, 1024, 1024);
}
